// MoEClusteredAttention_43035572305977
// MI455X (gfx1250) — matrix operand order repaired and checked by reading, not compiled, not run on hardware
//
#include <hip/hip_runtime.h>
#include <math.h>

#define BB 4
#define SQ 2048
#define SK 2048
#define DD 512
#define MM 8
#define NTOK (BB * (SQ + SK))
#define SCALE 0.044194173824159216f  // 1/sqrt(512)

#define LROW 520  // bf16 elements per padded LDS row (1024B data + 16B pad)

typedef __bf16 bf16;
typedef __attribute__((ext_vector_type(16))) __bf16 v16bf;
typedef __attribute__((ext_vector_type(8)))  __bf16 v8bf;
typedef __attribute__((ext_vector_type(8)))  float  v8f;
typedef unsigned int uint;
typedef uint u32x4 __attribute__((ext_vector_type(4)));
typedef uint u32x8 __attribute__((ext_vector_type(8)));

#define AS3 __attribute__((address_space(3)))

__device__ __forceinline__ v8f zero8() {
    v8f z;
#pragma unroll
    for (int i = 0; i < 8; ++i) z[i] = 0.0f;
    return z;
}

__device__ __forceinline__ v8f wmma_bf16(v16bf a, v16bf b, v8f c) {
    return __builtin_amdgcn_wmma_f32_16x16x32_bf16(false, a, false, b, (short)0, c,
                                                   false, false);
}

// A-matrix 16x32 bf16 fragment: per lane 16 contiguous K = {2*half..2*half+15}.
__device__ __forceinline__ v16bf load_a_bf16(const bf16* row, int half) {
    const v8bf lo = *reinterpret_cast<const v8bf*>(row + 2 * half);
    const v8bf hi = *reinterpret_cast<const v8bf*>(row + 2 * half + 8);
    v16bf r;
#pragma unroll
    for (int i = 0; i < 8; ++i) { r[i] = lo[i]; r[8 + i] = hi[i]; }
    return r;
}

// B-matrix 32x16 bf16 fragment: per lane 16 contiguous K; built from two 16B loads
// (16B alignment only — LDS rows are padded to 1040B).
__device__ __forceinline__ v16bf load_b16(const bf16* p) {
    const v8bf lo = *reinterpret_cast<const v8bf*>(p);
    const v8bf hi = *reinterpret_cast<const v8bf*>(p + 8);
    v16bf r;
#pragma unroll
    for (int i = 0; i < 8; ++i) { r[i] = lo[i]; r[8 + i] = hi[i]; }
    return r;
}

__device__ __forceinline__ v16bf load_a_f32(const float* row, int half) {
    v16bf r;
#pragma unroll
    for (int i = 0; i < 8; ++i) {
        r[i]     = (bf16)row[2 * half + i];
        r[8 + i] = (bf16)row[2 * half + 8 + i];
    }
    return r;
}

// --------------------------- Tensor Data Mover (ISA 08_async_tensor §7-8) ---
// D# group 0: count=1, lds_addr, 57-bit global addr, type=2.
__device__ __forceinline__ u32x4 tdm_g0(uint lds, unsigned long long ga) {
    u32x4 g;
    g[0] = 1u;
    g[1] = lds;
    g[2] = (uint)ga;
    g[3] = (uint)((ga >> 32) & 0x01FFFFFFull) | (2u << 30);
    return g;
}
// D# group 1: data_size=8B, pad 4 dwords every 256 dwords (1024B rows -> 1040B in LDS),
// 2-D tile: tile_dim0=128 (x8B = row), tile_dim1=rows, stride0=128.
__device__ __forceinline__ u32x8 tdm_g1(uint rows) {
    u32x8 g;
    const uint td0 = 128u, st0 = 128u, tile0 = 128u, td1 = 1u << 20;
    g[0] = (3u << 16) | (1u << 20) | (7u << 22) | (3u << 25);
    g[1] = (td0 & 0xffffu) << 16;
    g[2] = (td0 >> 16) | ((td1 & 0xffffu) << 16);
    g[3] = (td1 >> 16) | (tile0 << 16);
    g[4] = rows;  // tile_dim1; tile_dim2 = 0 (2-D)
    g[5] = st0;
    g[6] = 0u;
    g[7] = 0u;
    return g;
}

__device__ __forceinline__ void tdm_load(u32x4 g0, u32x8 g1) {
    asm volatile("tensor_load_to_lds %0, %1" ::"s"(g0), "s"(g1) : "memory");
}

__device__ __forceinline__ uint lds_offset(const void* p) {
    return (uint)(unsigned long long)(AS3 const char*)p;
}

// ---------------------------------------------------------------- routing ---
__global__ __launch_bounds__(256) void route_kernel(const float* __restrict__ Q,
                                                    const float* __restrict__ K,
                                                    const float* __restrict__ miu,
                                                    int* __restrict__ assign) {
    __shared__ float sm[MM * DD];
    for (int i = threadIdx.x; i < MM * DD; i += 256) sm[i] = miu[i];
    __syncthreads();
    const int wave = threadIdx.x >> 5, lane = threadIdx.x & 31;
    const int t = blockIdx.x * 8 + wave;
    if (t >= NTOK) return;
    const int b = t / (SQ + SK), pos = t % (SQ + SK);
    const float* x = (pos < SQ) ? Q + ((size_t)b * SQ + pos) * DD
                                : K + ((size_t)b * SK + (pos - SQ)) * DD;
    const int base = lane * 16;
    float xv[16];
#pragma unroll
    for (int i = 0; i < 16; ++i) xv[i] = x[base + i];
    float acc[MM];
#pragma unroll
    for (int m = 0; m < MM; ++m) {
        float a = 0.0f;
#pragma unroll
        for (int i = 0; i < 16; ++i) a += xv[i] * sm[m * DD + base + i];
        acc[m] = a;
    }
#pragma unroll
    for (int m = 0; m < MM; ++m)
        for (int off = 16; off > 0; off >>= 1) acc[m] += __shfl_xor(acc[m], off);
    if (lane == 0) {
        int best = 0; float bv = acc[0];
#pragma unroll
        for (int m = 1; m < MM; ++m)  // strict > keeps first-max (jnp.argmax)
            if (acc[m] > bv) { bv = acc[m]; best = m; }
        assign[t] = best;
    }
}

// ------------------------------------------------------------- compaction ---
__global__ void zero_counts_kernel(int* cq, int* ck) {
    int i = threadIdx.x;
    if (i < BB * MM) { cq[i] = 0; ck[i] = 0; }
}

__global__ void compact_kernel(const int* __restrict__ assign, int* cq, int* ck,
                               int* __restrict__ lq, int* __restrict__ lk) {
    int t = blockIdx.x * 256 + threadIdx.x;
    if (t >= NTOK) return;
    int b = t / (SQ + SK), pos = t % (SQ + SK), m = assign[t];
    if (pos < SQ) {
        int s = atomicAdd(&cq[b * MM + m], 1);
        lq[((size_t)(b * MM + m)) * SQ + s] = pos;
    } else {
        int s = atomicAdd(&ck[b * MM + m], 1);
        lk[((size_t)(b * MM + m)) * SK + s] = pos - SQ;
    }
}

// ---------------------------------------------- weight pack (B-frag order) ---
__global__ void pack_w_kernel(const float* __restrict__ W, bf16* __restrict__ Wp) {
    size_t idx = (size_t)blockIdx.x * 256 + threadIdx.x;
    if (idx >= (size_t)MM * DD * DD) return;
    int i  = idx & 15;
    int ln = (idx >> 4) & 31;
    size_t f = idx >> 9;
    int nt = f % (DD / 16);
    size_t f2 = f / (DD / 16);
    int kt = f2 % (DD / 32);
    int mm = f2 / (DD / 32);
    int k = kt * 32 + ((ln >> 4) & 1) * 16 + i;
    int n = nt * 16 + (ln & 15);
    Wp[idx] = (bf16)W[((size_t)mm * DD + k) * DD + n];
}

// ------------------------------------------------------- expert projection ---
__global__ __launch_bounds__(128) void proj_kernel(
    const float* __restrict__ X, const bf16* __restrict__ Wpack,
    const float* __restrict__ bias, const int* __restrict__ counts,
    const int* __restrict__ lists, bf16* __restrict__ Yp, int S) {
    const int lane = threadIdx.x & 31, wave = threadIdx.x >> 5;
    const int col = lane & 15, hi = (lane >> 4) & 1, half = hi * 8;
    const int tiles = S / 16;
    const int tile = blockIdx.x % tiles;
    const int m = (blockIdx.x / tiles) % MM;
    const int b = blockIdx.x / (tiles * MM);
    const int cnt = counts[b * MM + m];
    const int base = tile * 16;
    if (base >= cnt) return;  // uniform per block
    const int* lst = lists + ((size_t)(b * MM + m)) * S;
    const int nvalid = min(16, cnt - base);
    const int tokA = lst[base + min(col, nvalid - 1)];
    const float* xrow = X + ((size_t)b * S + tokA) * DD;
    const bf16* wbase = Wpack + (size_t)m * (DD / 32) * (DD / 16) * 512;
    const int ntBase = wave * 8;

    v8f acc[8];
#pragma unroll
    for (int j = 0; j < 8; ++j) acc[j] = zero8();

    auto loadB = [&](int kt, int j) {
        return *reinterpret_cast<const v16bf*>(
            wbase + ((size_t)kt * (DD / 16) + ntBase + j) * 512 + lane * 16);
    };

    // one-deep software pipeline: fragment j+1 (or next kt) loads overlap WMMA j
    v16bf aF = load_a_f32(xrow, half);
    v16bf bF = loadB(0, 0);
    for (int kt = 0; kt < DD / 32; ++kt) {
        v16bf aN = aF;
        if (kt + 1 < DD / 32) aN = load_a_f32(xrow + (kt + 1) * 32, half);
#pragma unroll
        for (int j = 0; j < 8; ++j) {
            v16bf bN = bF;
            if (j < 7) bN = loadB(kt, j + 1);
            else if (kt + 1 < DD / 32) bN = loadB(kt + 1, 0);
            acc[j] = wmma_bf16(aF, bF, acc[j]);
            bF = bN;
        }
        aF = aN;
    }
#pragma unroll
    for (int j = 0; j < 8; ++j) {
        const int n = (ntBase + j) * 16 + col;
        const float bv = bias[m * DD + n];
#pragma unroll
        for (int v = 0; v < 8; ++v) {
            const int r = v + half;
            if (r < nvalid) {
                const int tok = lst[base + r];
                float x = acc[j][v] + bv;
                float g = 0.5f * x * (1.0f + erff(x * 0.70710678118654752f));
                Yp[((size_t)b * S + tok) * DD + n] = (bf16)g;
            }
        }
    }
}

// -------------------------------------------------- Kp transpose + mean(V) ---
__global__ void transpose_kernel(const bf16* __restrict__ Kp, bf16* __restrict__ KpT) {
    size_t idx = (size_t)blockIdx.x * 256 + threadIdx.x;  // output-linear
    if (idx >= (size_t)BB * DD * SK) return;
    int s = idx % SK;
    size_t r = idx / SK;
    int d = r % DD;
    int b = r / DD;
    KpT[idx] = Kp[((size_t)b * SK + s) * DD + d];
}

__global__ void meanv_kernel(const bf16* __restrict__ Kp, float* __restrict__ meanV) {
    int idx = blockIdx.x * 256 + threadIdx.x;
    if (idx >= BB * DD) return;
    int d = idx % DD, b = idx / DD;
    float acc = 0.0f;
    for (int s = 0; s < SK; ++s) acc += (float)Kp[((size_t)b * SK + s) * DD + d];
    meanV[idx] = acc * (1.0f / SK);
}

// ------------------------------------------------------- masked attention ---
__global__ __launch_bounds__(128) void attn_kernel(
    const bf16* __restrict__ Qp, const bf16* __restrict__ Kp,
    const bf16* __restrict__ KpT, const int* __restrict__ assign,
    const float* __restrict__ meanV, const float* __restrict__ Q,
    float* __restrict__ out) {
    __shared__ bf16 qtile[16 * LROW];      // 16 query rows, TDM-staged
    __shared__ bf16 ktile[32 * LROW];      // 32 key rows, TDM-staged
    __shared__ bf16 Plds[4][16][32];       // per-wave P transpose tile

    const int lane = threadIdx.x & 31, wave = threadIdx.x >> 5;
    const int col = lane & 15, hi = (lane >> 4) & 1, half = hi * 8;
    const int qt = blockIdx.x % (SQ / 16);
    const int b  = blockIdx.x / (SQ / 16);
    const int dBase = wave * 128;  // 8 n-tiles of 16 dims per wave

    const int* aq = assign + (size_t)b * (SQ + SK);
    const int* ak = aq + SQ;

    int aqr[8];
#pragma unroll
    for (int v = 0; v < 8; ++v) aqr[v] = aq[qt * 16 + v + half];

    float mrow[8], lrow[8];
#pragma unroll
    for (int v = 0; v < 8; ++v) { mrow[v] = -1e30f; lrow[v] = 0.0f; }
    v8f Oacc[8];
#pragma unroll
    for (int nt = 0; nt < 8; ++nt) Oacc[nt] = zero8();

    const u32x8 g1q = tdm_g1(16u);
    const u32x8 g1k = tdm_g1(32u);
    const unsigned long long gaQ =
        (unsigned long long)(const void*)(Qp + ((size_t)b * SQ + qt * 16) * DD);
    const unsigned long long gaK0 =
        (unsigned long long)(const void*)(Kp + (size_t)b * SK * DD);

    if (wave == 0)  // TDM: stage the 16x512 Qp tile once per block
        tdm_load(tdm_g0(lds_offset(qtile), gaQ), g1q);

    const bf16* aRowL = qtile + col * LROW;          // A-frag row (query = col)
    const bf16* bRow0L = ktile + col * LROW;         // B-frag rows (keys)
    const bf16* bRow1L = ktile + (16 + col) * LROW;

    for (int kp = 0; kp < SK / 32; ++kp) {
        __syncthreads();  // all waves done reading previous key tile
        if (wave == 0) {  // TDM: stage 32x512 Kp tile, wait on TENSORcnt
            tdm_load(tdm_g0(lds_offset(ktile), gaK0 + (size_t)kp * 32 * DD * 2), g1k);
            __builtin_amdgcn_s_wait_tensorcnt(0);
        }
        __syncthreads();

        v8f S0 = zero8(), S1 = zero8();
#pragma unroll 4
        for (int kt = 0; kt < DD / 32; ++kt) {
            v16bf aF = load_a_bf16(aRowL + kt * 32, half);
            v16bf b0 = load_b16(bRow0L + kt * 32 + hi * 16);
            v16bf b1 = load_b16(bRow1L + kt * 32 + hi * 16);
            S0 = wmma_bf16(aF, b0, S0);
            S1 = wmma_bf16(aF, b1, S1);
        }
        const int key0 = kp * 32 + col, key1 = key0 + 16;
        const int ak0 = ak[key0], ak1 = ak[key1];
        float corrv[8];
#pragma unroll
        for (int v = 0; v < 8; ++v) {
            const bool m0 = (aqr[v] == ak0), m1 = (aqr[v] == ak1);
            const float s0 = S0[v] * SCALE, s1 = S1[v] * SCALE;
            float t = fmaxf(m0 ? s0 : -1e30f, m1 ? s1 : -1e30f);
            t = fmaxf(t, __shfl_xor(t, 1));
            t = fmaxf(t, __shfl_xor(t, 2));
            t = fmaxf(t, __shfl_xor(t, 4));
            t = fmaxf(t, __shfl_xor(t, 8));
            const float nm = fmaxf(mrow[v], t);
            const float corr = __expf(mrow[v] - nm);
            const float p0 = m0 ? __expf(s0 - nm) : 0.0f;
            const float p1 = m1 ? __expf(s1 - nm) : 0.0f;
            float ps = p0 + p1;
            ps += __shfl_xor(ps, 1);
            ps += __shfl_xor(ps, 2);
            ps += __shfl_xor(ps, 4);
            ps += __shfl_xor(ps, 8);
            lrow[v] = lrow[v] * corr + ps;
            mrow[v] = nm;
            corrv[v] = corr;
            Plds[wave][v + half][col]      = (bf16)p0;
            Plds[wave][v + half][16 + col] = (bf16)p1;
        }
#pragma unroll
        for (int nt = 0; nt < 8; ++nt)
#pragma unroll
            for (int v = 0; v < 8; ++v) Oacc[nt][v] *= corrv[v];

        asm volatile("s_wait_dscnt 0" ::: "memory");  // P stores -> A-frag loads

        const v16bf aP = load_a_bf16(&Plds[wave][col][0], half);
        const bf16* ktbase = KpT + ((size_t)b * DD + dBase + col) * SK + kp * 32 + hi * 16;
        if (kp + 1 < SK / 32)  // prefetch next KpT block (global_prefetch_b8)
            __builtin_prefetch(ktbase + 32, 0, 1);
        v16bf bF = *reinterpret_cast<const v16bf*>(ktbase);
#pragma unroll
        for (int nt = 0; nt < 8; ++nt) {
            v16bf bN = bF;
            if (nt + 1 < 8)
                bN = *reinterpret_cast<const v16bf*>(ktbase + (size_t)(nt + 1) * 16 * SK);
            Oacc[nt] = wmma_bf16(aP, bF, Oacc[nt]);
            bF = bN;
        }
    }

#pragma unroll
    for (int nt = 0; nt < 8; ++nt) {
        const int dim = dBase + nt * 16 + col;
#pragma unroll
        for (int v = 0; v < 8; ++v) {
            const int row = qt * 16 + v + half;
            const size_t o = ((size_t)b * SQ + row) * DD + dim;
            const float val = (lrow[v] > 0.0f) ? Oacc[nt][v] / lrow[v]
                                               : meanV[b * DD + dim];
            out[o] = val + Q[o];
        }
    }
}

// -------------------------------------------------------------- launcher ----
extern "C" void kernel_launch(void* const* d_in, const int* in_sizes, int n_in,
                              void* d_out, int out_size, void* d_ws, size_t ws_size,
                              hipStream_t stream) {
    const float* Q   = (const float*)d_in[0];
    const float* K   = (const float*)d_in[1];
    // d_in[2] = V (unused: Vp = Kp in the reference)
    const float* miu = (const float*)d_in[3];
    const float* Wq  = (const float*)d_in[4];
    const float* bq  = (const float*)d_in[5];
    const float* Wk  = (const float*)d_in[6];
    const float* bk  = (const float*)d_in[7];
    float* out = (float*)d_out;

    char* ws = (char*)d_ws;
    size_t off = 0;
    auto alloc = [&](size_t bytes) {
        size_t p = off;
        off += (bytes + 255) & ~(size_t)255;
        return p;
    };
    int*  assign = (int*)(ws + alloc((size_t)NTOK * 4));
    int*  cq     = (int*)(ws + alloc(BB * MM * 4));
    int*  ck     = (int*)(ws + alloc(BB * MM * 4));
    int*  lq     = (int*)(ws + alloc((size_t)BB * MM * SQ * 4));
    int*  lk     = (int*)(ws + alloc((size_t)BB * MM * SK * 4));
    bf16* Wqp    = (bf16*)(ws + alloc((size_t)MM * DD * DD * 2));
    bf16* Wkp    = (bf16*)(ws + alloc((size_t)MM * DD * DD * 2));
    bf16* Qp     = (bf16*)(ws + alloc((size_t)BB * SQ * DD * 2));
    bf16* Kp     = (bf16*)(ws + alloc((size_t)BB * SK * DD * 2));
    bf16* KpT    = (bf16*)(ws + alloc((size_t)BB * DD * SK * 2));
    float* meanV = (float*)(ws + alloc((size_t)BB * DD * 4));

    zero_counts_kernel<<<1, 64, 0, stream>>>(cq, ck);
    route_kernel<<<NTOK / 8, 256, 0, stream>>>(Q, K, miu, assign);
    compact_kernel<<<(NTOK + 255) / 256, 256, 0, stream>>>(assign, cq, ck, lq, lk);

    const int wElems = MM * DD * DD;
    pack_w_kernel<<<(wElems + 255) / 256, 256, 0, stream>>>(Wq, Wqp);
    pack_w_kernel<<<(wElems + 255) / 256, 256, 0, stream>>>(Wk, Wkp);

    proj_kernel<<<BB * MM * (SQ / 16), 128, 0, stream>>>(Q, Wqp, bq, cq, lq, Qp, SQ);
    proj_kernel<<<BB * MM * (SK / 16), 128, 0, stream>>>(K, Wkp, bk, ck, lk, Kp, SK);

    transpose_kernel<<<(int)(((size_t)BB * DD * SK + 255) / 256), 256, 0, stream>>>(Kp, KpT);
    meanv_kernel<<<(BB * DD + 255) / 256, 256, 0, stream>>>(Kp, meanV);

    attn_kernel<<<BB * (SQ / 16), 128, 0, stream>>>(Qp, Kp, KpT, assign, meanV, Q, out);
}
